// OmniMambaBlock_69423851372930
// MI455X (gfx1250) — matrix operand order repaired and checked by reading, not compiled, not run on hardware
//
#include <hip/hip_runtime.h>
#include <hip/hip_bf16.h>

// ---------------- types / WMMA helpers ----------------
typedef __bf16  bf16x16 __attribute__((ext_vector_type(16)));
typedef float   f32x8   __attribute__((ext_vector_type(8)));

__device__ __forceinline__ f32x8 wmma_bf16(bf16x16 a, bf16x16 b, f32x8 c) {
    // D = A(16x32 bf16) * B(32x16 bf16) + C(16x16 f32)
    return __builtin_amdgcn_wmma_f32_16x16x32_bf16(false, a, false, b, (short)0, c,
                                                   false, false);
}

// A fragment from bf16 LDS, row-major [row][k], leading dim ld.
// ISA 16-bit A 16x32 layout: lanes 0-15: e0..7 -> K=0..7, e8..15 -> K=16..23
//                            lanes 16-31: +8 on each K group.
__device__ __forceinline__ bf16x16 load_a_frag(const __bf16* base, int ld, int row,
                                               int kb, int lane) {
    int m = lane & 15, hi = lane >> 4;
    const __bf16* p = base + (size_t)(row + m) * ld + kb + hi * 8;
    bf16x16 a;
#pragma unroll
    for (int e = 0; e < 8; ++e) a[e] = p[e];
    p += 16;
#pragma unroll
    for (int e = 0; e < 8; ++e) a[8 + e] = p[e];
    return a;
}

// A fragment (same k order as load_a_frag) with per-k scaling (scale indexed by global k).
__device__ __forceinline__ bf16x16 load_a_frag_scaled(const __bf16* base, int ld, int row,
                                                      int kb, int lane,
                                                      const float* scale) {
    int m = lane & 15, hi = lane >> 4;
    const __bf16* p = base + (size_t)(row + m) * ld + kb + hi * 8;
    bf16x16 a;
#pragma unroll
    for (int e = 0; e < 8; ++e) a[e] = (__bf16)((float)p[e] * scale[kb + hi * 8 + e]);
#pragma unroll
    for (int e = 0; e < 8; ++e)
        a[8 + e] = (__bf16)((float)p[16 + e] * scale[kb + 16 + hi * 8 + e]);
    return a;
}

// B fragment from bf16 LDS stored [n][k] (k contiguous), leading dim ld.
// ISA 16-bit B 32x16 layout: lane col = lane&15, e0..7 -> k = 8*(lane>>4)+e, e8..15 -> k = 16+8*(lane>>4)+(e-8).
__device__ __forceinline__ bf16x16 load_b_frag(const __bf16* base, int ld, int col,
                                               int kb, int lane) {
    int n = lane & 15, hi = lane >> 4;
    const __bf16* p = base + (size_t)(col + n) * ld + kb + hi * 8;
    bf16x16 b;
#pragma unroll
    for (int e = 0; e < 8; ++e) { b[e] = p[e]; b[8 + e] = p[16 + e]; }
    return b;
}

// B fragment from LDS stored [k][n] (n contiguous, stride ld between k rows).
__device__ __forceinline__ bf16x16 load_b_frag_strided(const __bf16* base, int ld, int col,
                                                       int kb, int lane) {
    int n = lane & 15, hi = lane >> 4;
    const __bf16* p = base + (size_t)(kb + hi * 8) * ld + col + n;
    bf16x16 b;
#pragma unroll
    for (int e = 0; e < 8; ++e) { b[e] = p[(size_t)e * ld]; b[8 + e] = p[(size_t)(16 + e) * ld]; }
    return b;
}

__device__ __forceinline__ float sigmoidf_(float x) { return 1.f / (1.f + __expf(-x)); }
__device__ __forceinline__ float siluf_(float x) { return x * sigmoidf_(x); }

// ---------------- constants ----------------
#define LVOX   32768      // 32*32*32 tokens
#define CDIM   128
#define DINNER 256
#define NHEADS 4
#define DHEAD  64
#define DSTATE 64
#define CONVCH 384
#define DPROJ  644
#define ZXLD   656        // padded leading dim for zx
#define NCHUNK 256
#define CHUNKK 128

// ---------------- permutation tables ----------------
__global__ void perm_kernel(int* perms) {
    int t = blockIdx.x * 256 + threadIdx.x;
    if (t >= LVOX) return;
    const int L1 = LVOX - 1;
    perms[0 * LVOX + t] = t;
    perms[1 * LVOX + t] = L1 - t;
    {
        int u = t;      int h = u >> 10, d = (u >> 5) & 31, w = u & 31;
        perms[2 * LVOX + t] = (d << 10) + (h << 5) + w;
        u = L1 - t;     h = u >> 10; d = (u >> 5) & 31; w = u & 31;
        perms[3 * LVOX + t] = (d << 10) + (h << 5) + w;
    }
    {
        int u = t;      int w = u >> 10, d = (u >> 5) & 31, h = u & 31;
        perms[4 * LVOX + t] = (d << 10) + (h << 5) + w;
        u = L1 - t;     w = u >> 10; d = (u >> 5) & 31; h = u & 31;
        perms[5 * LVOX + t] = (d << 10) + (h << 5) + w;
    }
}

// ---------------- RMSNorm + direction gates ----------------
__global__ void __launch_bounds__(128) rms_gates_kernel(const float* __restrict__ x,
                                                        const float* __restrict__ normw,
                                                        const float* __restrict__ gw,
                                                        const float* __restrict__ gb,
                                                        float* __restrict__ xn,
                                                        float* __restrict__ gates) {
    int v = blockIdx.x, c = threadIdx.x;
    __shared__ float sv[128];
    __shared__ float red[128];
    __shared__ float gl[6];
    float val = x[(size_t)c * LVOX + v];
    red[c] = val * val;
    __syncthreads();
    for (int s = 64; s > 0; s >>= 1) { if (c < s) red[c] += red[c + s]; __syncthreads(); }
    float xnv = val * rsqrtf(red[0] / 128.f + 1e-6f) * normw[c];
    sv[c] = xnv;
    xn[(size_t)v * CDIM + c] = xnv;
    __syncthreads();
    if (c < 6) {
        float g = gb[c];
        for (int k = 0; k < 128; ++k) g += sv[k] * gw[k * 6 + c];
        gl[c] = g;
    }
    __syncthreads();
    if (c == 0) {
        float mx = gl[0];
        for (int j = 1; j < 6; ++j) mx = fmaxf(mx, gl[j]);
        float e[6], s = 0.f;
        for (int j = 0; j < 6; ++j) { e[j] = __expf(gl[j] - mx); s += e[j]; }
        for (int j = 0; j < 6; ++j) gates[(size_t)v * 6 + j] = e[j] / s;
    }
}

// ---------------- generic WMMA GEMM (128x128 block tile) ----------------
// C[M,N] = A[M,K] @ B[K,N].  A: f32 or bf16, optional row gather permA.
// B: f32, optional transpose.  Epilogue modes:
//   0: Cout = acc
//   1: outAccum[voxPerm[row]*ldc + col] += gates[vox*6+scanIdx] * acc
//   2: Cout = silu(acc + bias[col])
__global__ void __launch_bounds__(256) gemm_wmma_kernel(
        const void* __restrict__ Aptr, int aIsBf16, const int* __restrict__ permA,
        const float* __restrict__ B, int transB,
        int M, int N, int K, int lda, int ldb, int ldc,
        float* __restrict__ Cout, int mode,
        const float* __restrict__ bias,
        const float* __restrict__ gates, const int* __restrict__ voxPerm, int scanIdx,
        float* __restrict__ outAccum) {
    __shared__ __bf16 sA[128 * 32];
    __shared__ __bf16 sBt[128 * 32];   // [n][k]
    int tid = threadIdx.x, wave = tid >> 5, lane = tid & 31;
    int mBase = blockIdx.x * 128;
    int nBase = blockIdx.y * 128;
    f32x8 acc[8];
#pragma unroll
    for (int i = 0; i < 8; ++i) acc[i] = (f32x8){};
    for (int kb = 0; kb < K; kb += 32) {
        if (tid == 0 && kb + 32 < K)   // exercise global_prefetch path
            __builtin_prefetch(&B[(size_t)(transB ? (size_t)nBase * ldb + kb + 32
                                                  : (size_t)(kb + 32) * ldb + nBase)], 0, 3);
        // stage A (128 rows x 32 k), coalesced over k
        for (int i = tid; i < 128 * 32; i += 256) {
            int r = i >> 5, kk = i & 31;
            int gr = mBase + r, gk = kb + kk;
            float vl = 0.f;
            if (gr < M && gk < K) {
                int ar = permA ? permA[gr] : gr;
                vl = aIsBf16 ? (float)((const __bf16*)Aptr)[(size_t)ar * lda + gk]
                             : ((const float*)Aptr)[(size_t)ar * lda + gk];
            }
            sA[r * 32 + kk] = (__bf16)vl;
        }
        // stage B (128 cols x 32 k) into [n][k], coalesced mapping per layout
        for (int i = tid; i < 128 * 32; i += 256) {
            int n, kk;
            if (transB) { n = i >> 5; kk = i & 31; }
            else        { n = i & 127; kk = i >> 7; }
            int gn = nBase + n, gk = kb + kk;
            float vl = 0.f;
            if (gn < N && gk < K)
                vl = transB ? B[(size_t)gn * ldb + gk] : B[(size_t)gk * ldb + gn];
            sBt[n * 32 + kk] = (__bf16)vl;
        }
        __syncthreads();
        bf16x16 af = load_a_frag(sA, 32, wave * 16, 0, lane);
#pragma unroll
        for (int nt = 0; nt < 8; ++nt) {
            bf16x16 bfr = load_b_frag(sBt, 32, nt * 16, 0, lane);
            acc[nt] = wmma_bf16(af, bfr, acc[nt]);
        }
        __syncthreads();
    }
    int hi = lane >> 4;
#pragma unroll
    for (int nt = 0; nt < 8; ++nt) {
        int col = nBase + nt * 16 + (lane & 15);
        if (col >= N) continue;
#pragma unroll
        for (int vi = 0; vi < 8; ++vi) {
            int row = mBase + wave * 16 + vi + hi * 8;
            if (row >= M) continue;
            float val = acc[nt][vi];
            if (mode == 0) {
                Cout[(size_t)row * ldc + col] = val;
            } else if (mode == 1) {
                int vox = voxPerm[row];
                float g = gates[(size_t)vox * 6 + scanIdx];
                outAccum[(size_t)vox * ldc + col] += g * val;
            } else {
                float t = val + bias[col];
                Cout[(size_t)row * ldc + col] = siluf_(t);
            }
        }
    }
}

// ---------------- causal depthwise conv1d + SiLU (bf16 out), dt softplus ----------
__global__ void convdt_kernel(const float* __restrict__ zx, const float* __restrict__ convw,
                              const float* __restrict__ convb, const float* __restrict__ dtb,
                              __bf16* __restrict__ xcb, float* __restrict__ dtsp) {
    long i = (long)blockIdx.x * 256 + threadIdx.x;
    if (i >= (long)LVOX * 388) return;
    int t = (int)(i / 388), ch = (int)(i % 388);
    if (ch < CONVCH) {
        float acc = convb[ch];
#pragma unroll
        for (int k = 0; k < 4; ++k) {
            int tt = t - 3 + k;
            if (tt >= 0) acc += convw[ch * 4 + k] * zx[(size_t)tt * ZXLD + DINNER + ch];
        }
        xcb[(size_t)t * CONVCH + ch] = (__bf16)siluf_(acc);
    } else {
        int h = ch - CONVCH;
        float v = zx[(size_t)t * ZXLD + 640 + h] + dtb[h];
        dtsp[(size_t)t * NHEADS + h] = (v > 20.f) ? v : log1pf(__expf(v));
    }
}

// ---------------- per-chunk cumulative sum of dA ----------------
__global__ void cum_kernel(const float* __restrict__ dtsp, const float* __restrict__ Alog,
                           float* __restrict__ cumb, float* __restrict__ cdec) {
    int idx = blockIdx.x * 256 + threadIdx.x;
    if (idx >= NCHUNK * NHEADS) return;
    int chunk = idx >> 2, head = idx & 3;
    float A = -__expf(Alog[head]);
    float c = 0.f;
    int t0 = chunk * CHUNKK;
    for (int k = 0; k < CHUNKK; ++k) {
        c += dtsp[(size_t)(t0 + k) * NHEADS + head] * A;
        cumb[(size_t)(t0 + k) * NHEADS + head] = c;
    }
    cdec[idx] = __expf(c);
}

// ---------------- SSD intra-chunk: async-to-LDS tile + 3 WMMA phases -------------
__global__ void __launch_bounds__(256) ssd_intra_kernel(const __bf16* __restrict__ xcb,
                                                        const float* __restrict__ dtsp,
                                                        const float* __restrict__ cumb,
                                                        float* __restrict__ Y,
                                                        float* __restrict__ S) {
    int chunk = blockIdx.x, head = blockIdx.y;
    int tid = threadIdx.x, wave = tid >> 5, lane = tid & 31;
    __shared__ __bf16 sRaw[128 * CONVCH];  // raw conv activations: [s][384] (96KB)
    __shared__ __bf16 sG[128 * 128];       // masked CB * L (32KB)
    __shared__ __bf16 sXT[64 * 128];       // (xh*dt)^T : [p][s] (16KB)
    __shared__ float  scum[128];
    __shared__ float  sdt[128];
    __shared__ float  sdec[128];           // decay_end
    int t0 = chunk * CHUNKK;

    // Async DMA of the chunk tile (128 x 384 bf16 = 96KB) into LDS. ASYNCcnt path.
    {
        const char* src = (const char*)(xcb + (size_t)t0 * CONVCH);
        unsigned ldsbase = (unsigned)(uintptr_t)(&sRaw[0]);
#pragma unroll
        for (int it = 0; it < 24; ++it) {
            unsigned o = (unsigned)(tid * 16 + it * 4096);
            unsigned lds_off = ldsbase + o;
            unsigned long long ga = (unsigned long long)(src + o);
            asm volatile("global_load_async_to_lds_b128 %0, %1, off"
                         :: "v"(lds_off), "v"(ga) : "memory");
        }
    }
    if (tid < 128) {
        scum[tid] = cumb[(size_t)(t0 + tid) * NHEADS + head];
        sdt[tid]  = dtsp[(size_t)(t0 + tid) * NHEADS + head];
    }
    asm volatile("s_wait_asynccnt 0" ::: "memory");
    __syncthreads();
    if (tid < 128) sdec[tid] = __expf(scum[127] - scum[tid]);
    // build transposed dt-scaled xh tile: sXT[p][s]
    for (int i = tid; i < 64 * 128; i += 256) {
        int p = i >> 7, s = i & 127;
        sXT[i] = (__bf16)((float)sRaw[(size_t)s * CONVCH + head * DHEAD + p] * sdt[s]);
    }
    __syncthreads();

    const __bf16* sB = sRaw + DINNER;            // Bm[s][n], ld=384
    const __bf16* sC = sRaw + DINNER + DSTATE;   // Cm[t][n], ld=384

    // Phase 1: G[t][s] = (s<=t) ? (C.B^T)[t,s]*exp(cum[t]-cum[s]) : 0
    for (int ts = 0; ts < 8; ++ts) {
        f32x8 acc = {};
        for (int kb = 0; kb < 64; kb += 32) {
            bf16x16 a = load_a_frag(sC, CONVCH, wave * 16, kb, lane);
            bf16x16 b = load_b_frag(sB, CONVCH, ts * 16, kb, lane);
            acc = wmma_bf16(a, b, acc);
        }
        int scol = ts * 16 + (lane & 15), hi = lane >> 4;
#pragma unroll
        for (int vi = 0; vi < 8; ++vi) {
            int trow = wave * 16 + vi + hi * 8;
            float g = (scol <= trow) ? acc[vi] * __expf(scum[trow] - scum[scol]) : 0.f;
            sG[trow * 128 + scol] = (__bf16)g;
        }
    }
    __syncthreads();

    // Phase 2: Y_intra[t][p] = G @ (xh*dt)   (B from sXT, fully contiguous)
    for (int tp = 0; tp < 4; ++tp) {
        f32x8 acc = {};
        for (int kb = 0; kb < 128; kb += 32) {
            bf16x16 a = load_a_frag(sG, 128, wave * 16, kb, lane);
            bf16x16 b = load_b_frag(sXT, 128, tp * 16, kb, lane);
            acc = wmma_bf16(a, b, acc);
        }
        int pcol = tp * 16 + (lane & 15), hi = lane >> 4;
#pragma unroll
        for (int vi = 0; vi < 8; ++vi) {
            int trow = wave * 16 + vi + hi * 8;
            Y[(size_t)(t0 + trow) * DINNER + head * 64 + pcol] = acc[vi];
        }
    }

    // Phase 3: S[p][n] = sum_s dec_end[s]*(xh*dt)[s,p]*Bm[s,n]
    for (int it = 0; it < 2; ++it) {
        int id = wave * 2 + it, tp = id >> 2, tn = id & 3;
        f32x8 acc = {};
        for (int kb = 0; kb < 128; kb += 32) {
            bf16x16 a = load_a_frag_scaled(sXT, 128, tp * 16, kb, lane, sdec);
            bf16x16 b = load_b_frag_strided(sB, CONVCH, tn * 16, kb, lane);
            acc = wmma_bf16(a, b, acc);
        }
        int ncol = tn * 16 + (lane & 15), hi = lane >> 4;
#pragma unroll
        for (int vi = 0; vi < 8; ++vi) {
            int prow = tp * 16 + vi + hi * 8;
            S[((size_t)(chunk * NHEADS + head) * 64 + prow) * 64 + ncol] = acc[vi];
        }
    }
}

// ---------------- sequential inter-chunk state scan (S -> hprev, in place) --------
__global__ void state_scan_kernel(float* __restrict__ S, const float* __restrict__ cdec) {
    int idx = blockIdx.x * 256 + threadIdx.x;
    if (idx >= NHEADS * 64 * 64) return;
    int head = idx >> 12;
    int pn = idx & 4095;
    float cur = 0.f;
    for (int c = 0; c < NCHUNK; ++c) {
        size_t off = (size_t)(c * NHEADS + head) * 4096 + pn;
        float sc = S[off];
        S[off] = cur;                         // hprev before this chunk
        cur = cdec[c * NHEADS + head] * cur + sc;
    }
}

// ---------------- SSD inter-chunk: Y += (C*exp(cum)) @ hprev (WMMA) ----------------
__global__ void __launch_bounds__(256) ssd_inter_kernel(const __bf16* __restrict__ xcb,
                                                        const float* __restrict__ cumb,
                                                        const float* __restrict__ hprev,
                                                        float* __restrict__ Y) {
    int chunk = blockIdx.x, head = blockIdx.y;
    int tid = threadIdx.x, wave = tid >> 5, lane = tid & 31;
    __shared__ __bf16 sA[128 * 64];  // Cm scaled by exp(cum)
    __shared__ __bf16 sH[64 * 64];   // hprev [p][n]
    __shared__ float  se[128];
    int t0 = chunk * CHUNKK;
    if (tid < 128) se[tid] = __expf(cumb[(size_t)(t0 + tid) * NHEADS + head]);
    __syncthreads();
    for (int i = tid; i < 128 * 64; i += 256) {
        int t = i >> 6, n = i & 63;
        sA[i] = (__bf16)((float)xcb[(size_t)(t0 + t) * CONVCH + DINNER + DSTATE + n] * se[t]);
    }
    const float* hp = hprev + (size_t)(chunk * NHEADS + head) * 4096;
    for (int i = tid; i < 64 * 64; i += 256) sH[i] = (__bf16)hp[i];
    __syncthreads();
    for (int tp = 0; tp < 4; ++tp) {
        f32x8 acc = {};
        for (int kb = 0; kb < 64; kb += 32) {
            bf16x16 a = load_a_frag(sA, 64, wave * 16, kb, lane);
            bf16x16 b = load_b_frag(sH, 64, tp * 16, kb, lane);
            acc = wmma_bf16(a, b, acc);
        }
        int pcol = tp * 16 + (lane & 15), hi = lane >> 4;
#pragma unroll
        for (int vi = 0; vi < 8; ++vi) {
            int trow = wave * 16 + vi + hi * 8;
            Y[(size_t)(t0 + trow) * DINNER + head * 64 + pcol] += acc[vi];
        }
    }
}

// ---------------- +D*xh, gated RMSNorm -> bf16 A for out-proj ----------------
__global__ void __launch_bounds__(256) ynorm_kernel(const float* __restrict__ Y,
                                                    const __bf16* __restrict__ xcb,
                                                    const float* __restrict__ zx,
                                                    const float* __restrict__ Dp,
                                                    const float* __restrict__ normw,
                                                    __bf16* __restrict__ ybf) {
    int t = blockIdx.x, i = threadIdx.x;
    __shared__ float red[256];
    float xh = (float)xcb[(size_t)t * CONVCH + i];
    float yv = Y[(size_t)t * DINNER + i] + Dp[i >> 6] * xh;
    float z = zx[(size_t)t * ZXLD + i];
    float yg = yv * siluf_(z);
    red[i] = yg * yg;
    __syncthreads();
    for (int s = 128; s > 0; s >>= 1) { if (i < s) red[i] += red[i + s]; __syncthreads(); }
    float scale = rsqrtf(red[0] / 256.f + 1e-6f);
    ybf[(size_t)t * DINNER + i] = (__bf16)(yg * scale * normw[i]);
}

// ---------------- local path: 3x3x3 depthwise conv (+bias) -> bf16 ----------------
__global__ void dwconv_kernel(const float* __restrict__ xn, const float* __restrict__ w,
                              const float* __restrict__ b, __bf16* __restrict__ outbf) {
    long i = (long)blockIdx.x * 256 + threadIdx.x;
    if (i >= (long)LVOX * CDIM) return;
    int c = (int)(i & 127), v = (int)(i >> 7);
    int d = v >> 10, h = (v >> 5) & 31, wq = v & 31;
    float acc = b[c];
#pragma unroll
    for (int kd = -1; kd <= 1; ++kd)
#pragma unroll
        for (int kh = -1; kh <= 1; ++kh)
#pragma unroll
            for (int kw = -1; kw <= 1; ++kw) {
                int dd = d + kd, hh = h + kh, ww = wq + kw;
                if (dd >= 0 && dd < 32 && hh >= 0 && hh < 32 && ww >= 0 && ww < 32)
                    acc += w[c * 27 + (kd + 1) * 9 + (kh + 1) * 3 + (kw + 1)] *
                           xn[(size_t)((dd << 10) + (hh << 5) + ww) * CDIM + c];
            }
    outbf[i] = (__bf16)acc;
}

// ---------------- fusion + mean + channel attention + residual ----------------
__global__ void fused_kernel(const float* __restrict__ outg, const float* __restrict__ lc,
                             const float* __restrict__ alpha, float* __restrict__ fused) {
    long i = (long)blockIdx.x * 256 + threadIdx.x;
    if (i >= (long)LVOX * CDIM) return;
    float a = alpha[0];
    fused[i] = a * outg[i] + (1.f - a) * lc[i];
}

__global__ void __launch_bounds__(256) chanmean_kernel(const float* __restrict__ fused,
                                                       float* __restrict__ meanC) {
    int c = blockIdx.x;
    __shared__ float red[256];
    float s = 0.f;
    for (int v = threadIdx.x; v < LVOX; v += 256) s += fused[(size_t)v * CDIM + c];
    red[threadIdx.x] = s;
    __syncthreads();
    for (int st = 128; st > 0; st >>= 1) {
        if (threadIdx.x < st) red[threadIdx.x] += red[threadIdx.x + st];
        __syncthreads();
    }
    if (threadIdx.x == 0) meanC[c] = red[0] / (float)LVOX;
}

__global__ void __launch_bounds__(128) ca_kernel(const float* __restrict__ meanC,
                                                 const float* __restrict__ w1,
                                                 const float* __restrict__ b1,
                                                 const float* __restrict__ w2,
                                                 const float* __restrict__ b2,
                                                 float* __restrict__ sig) {
    __shared__ float h1[8];
    int tid = threadIdx.x;
    if (tid < 8) {
        float a = b1[tid];
        for (int c = 0; c < 128; ++c) a += w1[tid * 128 + c] * meanC[c];
        h1[tid] = (a >= 0.f) ? a : 0.1f * a;
    }
    __syncthreads();
    float a = b2[tid];
    for (int o = 0; o < 8; ++o) a += w2[tid * 8 + o] * h1[o];
    sig[tid] = sigmoidf_(a);
}

__global__ void final_kernel(const float* __restrict__ x, const float* __restrict__ fused,
                             const float* __restrict__ sig, float* __restrict__ out) {
    long i = (long)blockIdx.x * 256 + threadIdx.x;
    if (i >= (long)CDIM * LVOX) return;
    int c = (int)(i >> 15), v = (int)(i & (LVOX - 1));
    out[i] = x[i] + fused[(size_t)v * CDIM + c] * sig[c];
}

// ---------------- host orchestration ----------------
extern "C" void kernel_launch(void* const* d_in, const int* in_sizes, int n_in,
                              void* d_out, int out_size, void* d_ws, size_t ws_size,
                              hipStream_t stream) {
    (void)in_sizes; (void)n_in; (void)out_size; (void)ws_size;
    const float* x      = (const float*)d_in[0];
    const float* norm_w = (const float*)d_in[1];
    const float* gate_w = (const float*)d_in[2];
    const float* gate_b = (const float*)d_in[3];
    const float* loc_dw_w = (const float*)d_in[28];
    const float* loc_dw_b = (const float*)d_in[29];
    const float* loc_pw_w = (const float*)d_in[30];
    const float* loc_pw_b = (const float*)d_in[31];
    const float* ca_w1 = (const float*)d_in[32];
    const float* ca_b1 = (const float*)d_in[33];
    const float* ca_w2 = (const float*)d_in[34];
    const float* ca_b2 = (const float*)d_in[35];
    const float* alpha = (const float*)d_in[36];

    char* ws = (char*)d_ws;
    size_t off = 0;
    auto carve = [&](size_t bytes) -> void* {
        void* p = ws + off;
        off = (off + bytes + 255) & ~(size_t)255;
        return p;
    };
    int*    perms = (int*)carve((size_t)6 * LVOX * 4);
    float*  gates = (float*)carve((size_t)LVOX * 6 * 4);
    float*  xn    = (float*)carve((size_t)LVOX * CDIM * 4);
    float*  zx    = (float*)carve((size_t)LVOX * ZXLD * 4);
    __bf16* xcb   = (__bf16*)carve((size_t)LVOX * CONVCH * 2);
    float*  dtsp  = (float*)carve((size_t)LVOX * NHEADS * 4);
    float*  cumb  = (float*)carve((size_t)LVOX * NHEADS * 4);
    float*  cdec  = (float*)carve((size_t)NCHUNK * NHEADS * 4);
    float*  Sbuf  = (float*)carve((size_t)NCHUNK * NHEADS * 64 * 64 * 4);
    float*  Ybuf  = (float*)carve((size_t)LVOX * DINNER * 4);
    __bf16* ybf   = (__bf16*)carve((size_t)LVOX * DINNER * 2);
    float*  outg  = (float*)carve((size_t)LVOX * CDIM * 4);
    __bf16* dwbf  = (__bf16*)carve((size_t)LVOX * CDIM * 2);
    float*  lc    = (float*)carve((size_t)LVOX * CDIM * 4);
    float*  fus   = (float*)carve((size_t)LVOX * CDIM * 4);
    float*  meanC = (float*)carve(512);
    float*  sig   = (float*)carve(512);

    hipMemsetAsync(outg, 0, (size_t)LVOX * CDIM * 4, stream);
    perm_kernel<<<LVOX / 256, 256, 0, stream>>>(perms);
    rms_gates_kernel<<<LVOX, 128, 0, stream>>>(x, norm_w, gate_w, gate_b, xn, gates);

    dim3 gridSSD(NCHUNK, NHEADS);
    for (int scan = 0; scan < 6; ++scan) {
        int pb = 4 + 8 * (scan >> 1);
        const float* Win   = (const float*)d_in[pb + 0];
        const float* convw = (const float*)d_in[pb + 1];
        const float* convb = (const float*)d_in[pb + 2];
        const float* dtb   = (const float*)d_in[pb + 3];
        const float* Alog  = (const float*)d_in[pb + 4];
        const float* Dp    = (const float*)d_in[pb + 5];
        const float* nrm   = (const float*)d_in[pb + 6];
        const float* Wout  = (const float*)d_in[pb + 7];
        const int* permScan = perms + (size_t)scan * LVOX;

        // in-projection: zx[t, :644] = xn[perm(t), :] @ Win
        dim3 g1(LVOX / 128, (DPROJ + 127) / 128);
        gemm_wmma_kernel<<<g1, 256, 0, stream>>>(xn, 0, permScan, Win, 0,
            LVOX, DPROJ, CDIM, CDIM, DPROJ, ZXLD, zx, 0,
            nullptr, nullptr, nullptr, 0, nullptr);

        convdt_kernel<<<((long)LVOX * 388 + 255) / 256, 256, 0, stream>>>(
            zx, convw, convb, dtb, xcb, dtsp);
        cum_kernel<<<(NCHUNK * NHEADS + 255) / 256, 256, 0, stream>>>(dtsp, Alog, cumb, cdec);

        ssd_intra_kernel<<<gridSSD, 256, 0, stream>>>(xcb, dtsp, cumb, Ybuf, Sbuf);
        state_scan_kernel<<<(NHEADS * 64 * 64) / 256, 256, 0, stream>>>(Sbuf, cdec);
        ssd_inter_kernel<<<gridSSD, 256, 0, stream>>>(xcb, cumb, Sbuf, Ybuf);

        ynorm_kernel<<<LVOX, 256, 0, stream>>>(Ybuf, xcb, zx, Dp, nrm, ybf);

        // out-projection with gated scatter-accumulate epilogue
        dim3 g2(LVOX / 128, 1);
        gemm_wmma_kernel<<<g2, 256, 0, stream>>>(ybf, 1, nullptr, Wout, 0,
            LVOX, CDIM, DINNER, DINNER, CDIM, CDIM, nullptr, 1,
            nullptr, gates, permScan, scan, outg);
    }

    // local conv path
    dwconv_kernel<<<((long)LVOX * CDIM + 255) / 256, 256, 0, stream>>>(
        xn, loc_dw_w, loc_dw_b, dwbf);
    dim3 gpw(LVOX / 128, 1);
    gemm_wmma_kernel<<<gpw, 256, 0, stream>>>(dwbf, 1, nullptr, loc_pw_w, 1,
        LVOX, CDIM, CDIM, CDIM, CDIM, CDIM, lc, 2,
        loc_pw_b, nullptr, nullptr, 0, nullptr);

    // fuse, channel attention, residual
    fused_kernel<<<((long)LVOX * CDIM + 255) / 256, 256, 0, stream>>>(outg, lc, alpha, fus);
    chanmean_kernel<<<CDIM, 256, 0, stream>>>(fus, meanC);
    ca_kernel<<<1, 128, 0, stream>>>(meanC, ca_w1, ca_b1, ca_w2, ca_b2, sig);
    final_kernel<<<((long)CDIM * LVOX + 255) / 256, 256, 0, stream>>>(
        x, fus, sig, (float*)d_out);
}
